// AxialBlock_global_85873576116987
// MI455X (gfx1250) — matrix operand order repaired and checked by reading, not compiled, not run on hardware
//
#include <hip/hip_runtime.h>

// ---------------- problem constants ----------------
#define NB_   16
#define CIN_  128
#define COUT_ 256
#define MID_  128
#define G_    8
#define GP_   16
#define KK_   64
#define HW_   4096          // KK_*KK_
#define SEQB_ 1024          // NB_*KK_
#define EPS_  1e-5f

// ---------------- types for WMMA ----------------
typedef __attribute__((ext_vector_type(16))) __bf16 v16bf;
typedef __attribute__((ext_vector_type(8)))  float  v8f;

struct alignas(16) B16 { unsigned int x[4]; };

union FragAB {
  v16bf v;
  B16 q[2];
  unsigned short s[16];
};

// ---------------- helpers ----------------
__device__ __forceinline__ unsigned short f2bf(float f) {
  unsigned int u = __float_as_uint(f);
  u += 0x7FFFu + ((u >> 16) & 1u);
  return (unsigned short)(u >> 16);
}

__global__ void k_f32_to_bf16(const float* __restrict__ src,
                              unsigned short* __restrict__ dst, int n) {
  int i = blockIdx.x * blockDim.x + threadIdx.x;
  if (i < n) dst[i] = f2bf(src[i]);
}

__global__ void k_zero(float* __restrict__ p, int n) {
  int i = blockIdx.x * blockDim.x + threadIdx.x;
  if (i < n) p[i] = 0.0f;
}

// ---------------- tiled transpose+convert: x (n,c,hw) f32 -> xt (n,hw,c) bf16 ----------------
__global__ void __launch_bounds__(256)
k_xT(const float* __restrict__ x, unsigned short* __restrict__ xt)
{
  __shared__ unsigned short tile[32][33];
  int bid = blockIdx.x;            // n(16) * hwT(128) * cT(4)
  int cT  = bid & 3;
  int hwT = (bid >> 2) & 127;
  int nn  = bid >> 9;
  int tx = threadIdx.x & 31;
  int ty = threadIdx.x >> 5;       // 0..7
  #pragma unroll
  for (int p = 0; p < 4; ++p) {
    int c = cT * 32 + p * 8 + ty;
    tile[p * 8 + ty][tx] = f2bf(x[((nn * 128 + c) * 4096) + hwT * 32 + tx]);
  }
  __syncthreads();
  #pragma unroll
  for (int p = 0; p < 4; ++p) {
    int hw = hwT * 32 + p * 8 + ty;
    xt[((nn * 4096 + hw) * 128) + cT * 32 + tx] = tile[tx][p * 8 + ty];
  }
}

// ---------------- bf16 WMMA GEMM, K-major B (no LDS) ----------------
// A : M x Kd row-major bf16 (weights)
// Bt: ncols x Kd row-major bf16  (i.e. column-major B, k contiguous)
// C : (nbatch, M, SPW) f32, column cg = batch*SPW + sp
// One wave computes a 16(M) x 64(cols) strip: 4 accumulators.
__global__ void __launch_bounds__(128)
k_gemm_bf16(const unsigned short* __restrict__ A,
            const unsigned short* __restrict__ Bt,
            const float* __restrict__ bias,
            float* __restrict__ C,
            int M, int Kd, int SPW)
{
  const int wave = threadIdx.x >> 5;
  const int lane = threadIdx.x & 31;
  const int mTiles = M >> 4;
  const int wid    = blockIdx.x * 4 + wave;
  const int mTile  = wid % mTiles;
  const int cGroup = wid / mTiles;        // group of 64 columns
  const int colBase = cGroup * 64;
  const int n = lane & 15;
  const int h = lane >> 4;

  // A fragment: row m = mTile*16+n ; k runs: [k0+h*8, +8) and [k0+16+h*8, +8)
  const unsigned short* Arow = A + (mTile * 16 + n) * Kd + h * 8;
  // B fragment: col = colBase + t*16 + n ; k runs: [k0+h*8, +8) and [k0+16+h*8, +8)
  const unsigned short* Bcol = Bt + (long long)(colBase + n) * Kd + h * 8;

  v8f acc[4];
  #pragma unroll
  for (int t = 0; t < 4; ++t)
    #pragma unroll
    for (int r = 0; r < 8; ++r) acc[t][r] = 0.0f;

  for (int k0 = 0; k0 < Kd; k0 += 32) {
    FragAB a;
    a.q[0] = *(const B16*)(Arow + k0);
    a.q[1] = *(const B16*)(Arow + k0 + 16);
    #pragma unroll
    for (int t = 0; t < 4; ++t) {
      FragAB b;
      const unsigned short* bp = Bcol + (long long)t * 16 * Kd + k0;
      b.q[0] = *(const B16*)(bp);
      b.q[1] = *(const B16*)(bp + 16);
      acc[t] = __builtin_amdgcn_wmma_f32_16x16x32_bf16(
          false, a.v, false, b.v, (short)0, acc[t], false, false);
    }
  }

  const int batch = colBase / SPW;     // 64-col group never crosses batch (SPW % 64 == 0)
  const int sp0   = colBase % SPW;
  float* Cb = C + ((long long)batch * M) * SPW + sp0 + n;
  #pragma unroll
  for (int t = 0; t < 4; ++t) {
    #pragma unroll
    for (int r = 0; r < 8; ++r) {
      int m = mTile * 16 + r + h * 8;
      float v = acc[t][r] + (bias ? bias[m] : 0.0f);
      Cb[(long long)m * SPW + t * 16] = v;
    }
  }
}

// ---------------- per-channel stats (sum, sumsq) ----------------
__global__ void __launch_bounds__(256)
k_stats(const float* __restrict__ x, float* __restrict__ stats,
        int C, int nbatch, int S)
{
  int c = blockIdx.x;
  int tot = nbatch * S;
  float s = 0.f, s2 = 0.f;
  for (int t = threadIdx.x; t < tot; t += 256) {
    int b = t / S, sp = t - b * S;
    float v = x[((long long)b * C + c) * S + sp];
    s += v; s2 += v * v;
  }
  __shared__ float rs[256], rs2[256];
  rs[threadIdx.x] = s; rs2[threadIdx.x] = s2;
  __syncthreads();
  for (int off = 128; off > 0; off >>= 1) {
    if (threadIdx.x < off) {
      rs[threadIdx.x]  += rs[threadIdx.x + off];
      rs2[threadIdx.x] += rs2[threadIdx.x + off];
    }
    __syncthreads();
  }
  if (threadIdx.x == 0) { stats[c * 2] = rs[0]; stats[c * 2 + 1] = rs2[0]; }
}

__global__ void k_bn_finalize(const float* __restrict__ stats,
                              const float* __restrict__ g,
                              const float* __restrict__ b,
                              float* __restrict__ ss, int C, float invCount)
{
  int c = blockIdx.x * blockDim.x + threadIdx.x;
  if (c >= C) return;
  float mean = stats[c * 2] * invCount;
  float var  = stats[c * 2 + 1] * invCount - mean * mean;
  float sc   = g[c] * rsqrtf(var + EPS_);
  ss[c * 2]     = sc;
  ss[c * 2 + 1] = b[c] - mean * sc;
}

// ---------------- bn1 + relu, (n,c,h,w) -> h_inT [(n*64+w)*64+h][c] bf16 ----------------
__global__ void k_bn1_relu_to_hin(const float* __restrict__ mid,
                                  const float* __restrict__ ss,
                                  unsigned short* __restrict__ gin)
{
  int idx = blockIdx.x * blockDim.x + threadIdx.x;
  if (idx >= NB_ * MID_ * HW_) return;
  int w  = idx & 63;
  int hh = (idx >> 6) & 63;
  int c  = (idx >> 12) & 127;
  int nn = idx >> 19;
  float v = mid[idx] * ss[c * 2] + ss[c * 2 + 1];
  v = v > 0.f ? v : 0.f;
  gin[(((nn * 64 + w) * 64 + hh) * 128) + c] = f2bf(v);
}

// ---------------- attention stats pass: accumulate sim BN stats ----------------
__global__ void __launch_bounds__(256)
k_att_stats(const float* __restrict__ qkv, const float* __restrict__ ssq,
            const float* __restrict__ rel, float* __restrict__ simstats)
{
  int bg = blockIdx.x;
  int b = bg >> 3;
  int g = bg & 7;
  __shared__ float q[8][64], kmat[8][64];
  for (int t = threadIdx.x; t < 16 * 64; t += 256) {
    int c = t >> 6, i = t & 63;
    int ch = g * 32 + c;
    float v = qkv[((long long)b * 256 + ch) * 64 + i];
    v = v * ssq[ch * 2] + ssq[ch * 2 + 1];
    if (c < 8) q[c][i] = v; else kmat[c - 8][i] = v;
  }
  __syncthreads();
  float s0 = 0, s1 = 0, s2 = 0, s3 = 0, s4 = 0, s5 = 0;
  for (int t = threadIdx.x; t < 4096; t += 256) {
    int i = t >> 6, j = t & 63;
    int d1 = i - j + 63;
    int d2 = j - i + 63;
    float aqk = 0, aqr = 0, akr = 0;
    #pragma unroll
    for (int c = 0; c < 8; ++c) {
      aqk += q[c][i] * kmat[c][j];
      aqr += q[c][i] * rel[c * 127 + d1];
      akr += kmat[c][j] * rel[(8 + c) * 127 + d2];
    }
    aqr *= 0.1f; akr *= 0.1f;
    s0 += aqk; s1 += aqk * aqk;
    s2 += aqr; s3 += aqr * aqr;
    s4 += akr; s5 += akr * akr;
  }
  __shared__ float red[6][256];
  red[0][threadIdx.x] = s0; red[1][threadIdx.x] = s1; red[2][threadIdx.x] = s2;
  red[3][threadIdx.x] = s3; red[4][threadIdx.x] = s4; red[5][threadIdx.x] = s5;
  __syncthreads();
  for (int off = 128; off > 0; off >>= 1) {
    if (threadIdx.x < off)
      #pragma unroll
      for (int r = 0; r < 6; ++r)
        red[r][threadIdx.x] += red[r][threadIdx.x + off];
    __syncthreads();
  }
  if (threadIdx.x == 0) {
    atomicAdd(&simstats[g * 2 + 0],        red[0][0]);
    atomicAdd(&simstats[g * 2 + 1],        red[1][0]);
    atomicAdd(&simstats[(8 + g) * 2 + 0],  red[2][0]);
    atomicAdd(&simstats[(8 + g) * 2 + 1],  red[3][0]);
    atomicAdd(&simstats[(16 + g) * 2 + 0], red[4][0]);
    atomicAdd(&simstats[(16 + g) * 2 + 1], red[5][0]);
  }
}

// ---------------- attention apply: sim BN + softmax + sv/sve ----------------
__global__ void __launch_bounds__(64)
k_att_apply(const float* __restrict__ qkv, const float* __restrict__ ssq,
            const float* __restrict__ rel, const float* __restrict__ sssim,
            float* __restrict__ attout)
{
  int bg = blockIdx.x;
  int b = bg >> 3;
  int g = bg & 7;
  __shared__ float q[8][64], kmat[8][64], vmat[16][64];
  __shared__ float sm[64][65];
  for (int t = threadIdx.x; t < 32 * 64; t += 64) {
    int c = t >> 6, i = t & 63;
    int ch = g * 32 + c;
    float v = qkv[((long long)b * 256 + ch) * 64 + i];
    v = v * ssq[ch * 2] + ssq[ch * 2 + 1];
    if (c < 8) q[c][i] = v;
    else if (c < 16) kmat[c - 8][i] = v;
    else vmat[c - 16][i] = v;
  }
  __syncthreads();
  int i = threadIdx.x;
  float sc0 = sssim[g * 2],        sh0 = sssim[g * 2 + 1];
  float sc1 = sssim[(8 + g) * 2],  sh1 = sssim[(8 + g) * 2 + 1];
  float sc2 = sssim[(16 + g) * 2], sh2 = sssim[(16 + g) * 2 + 1];
  float mx = -3.4e38f;
  for (int j = 0; j < 64; ++j) {
    int d1 = i - j + 63, d2 = j - i + 63;
    float aqk = 0, aqr = 0, akr = 0;
    #pragma unroll
    for (int c = 0; c < 8; ++c) {
      aqk += q[c][i] * kmat[c][j];
      aqr += q[c][i] * rel[c * 127 + d1];
      akr += kmat[c][j] * rel[(8 + c) * 127 + d2];
    }
    float s = (aqk * sc0 + sh0) + (0.1f * aqr * sc1 + sh1) + (0.1f * akr * sc2 + sh2);
    sm[i][j] = s;
    mx = fmaxf(mx, s);
  }
  float sum = 0.f;
  for (int j = 0; j < 64; ++j) { float e = __expf(sm[i][j] - mx); sm[i][j] = e; sum += e; }
  float inv = 1.0f / sum;
  for (int j = 0; j < 64; ++j) sm[i][j] *= inv;
  for (int c = 0; c < 16; ++c) {
    float a = 0.f, ae = 0.f;
    const float* rv = rel + (16 + c) * 127;
    for (int j = 0; j < 64; ++j) {
      float s = sm[i][j];
      a  += s * vmat[c][j];
      ae += s * rv[i - j + 63];
    }
    attout[((long long)b * 256 + (g * 16 + c)) * 64 + i] = a;
    attout[((long long)b * 256 + (128 + g * 16 + c)) * 64 + i] = 0.1f * ae;
  }
}

// ---------------- bo BN + pair-sum, height -> w_inT [( (n*64+h)*64+w )][m] ----------------
__global__ void k_bo_apply_h(const float* __restrict__ att,
                             const float* __restrict__ ss,
                             unsigned short* __restrict__ gin)
{
  int idx = blockIdx.x * blockDim.x + threadIdx.x; // SEQB_*MID_*64
  if (idx >= SEQB_ * MID_ * 64) return;
  int i  = idx & 63;          // seq index (h)
  int m  = (idx >> 6) & 127;
  int b1 = idx >> 13;         // n*64 + w
  int w  = b1 & 63, nn = b1 >> 6;
  float v0 = att[((long long)b1 * 256 + 2 * m) * 64 + i] * ss[4 * m]         + ss[4 * m + 1];
  float v1 = att[((long long)b1 * 256 + 2 * m + 1) * 64 + i] * ss[4 * m + 2] + ss[4 * m + 3];
  gin[(((nn * 64 + i) * 64 + w) * 128) + m] = f2bf(v0 + v1);
}

// ---------------- bo BN + pair-sum, width -> cu_inT [(n*4096 + h*64 + w)][m] ----------------
__global__ void k_bo_apply_w(const float* __restrict__ att,
                             const float* __restrict__ ss,
                             unsigned short* __restrict__ gin)
{
  int idx = blockIdx.x * blockDim.x + threadIdx.x;
  if (idx >= SEQB_ * MID_ * 64) return;
  int j  = idx & 63;          // seq index (w)
  int m  = (idx >> 6) & 127;
  int b2 = idx >> 13;         // n*64 + h
  int hh = b2 & 63, nn = b2 >> 6;
  float v0 = att[((long long)b2 * 256 + 2 * m) * 64 + j] * ss[4 * m]         + ss[4 * m + 1];
  float v1 = att[((long long)b2 * 256 + 2 * m + 1) * 64 + j] * ss[4 * m + 2] + ss[4 * m + 3];
  gin[((nn * 4096 + hh * 64 + j) * 128) + m] = f2bf(v0 + v1);
}

// ---------------- bn2 + residual ----------------
__global__ void k_bn2_res(const float* __restrict__ y, const float* __restrict__ ss,
                          const float* __restrict__ xout, float* __restrict__ out)
{
  int idx = blockIdx.x * blockDim.x + threadIdx.x;
  if (idx >= NB_ * COUT_ * HW_) return;
  int c = (idx >> 12) & 255;
  out[idx] = y[idx] * ss[c * 2] + ss[c * 2 + 1] + xout[idx];
}

// ---------------- host launch ----------------
extern "C" void kernel_launch(void* const* d_in, const int* in_sizes, int n_in,
                              void* d_out, int out_size, void* d_ws, size_t ws_size,
                              hipStream_t stream) {
  (void)in_sizes; (void)n_in; (void)out_size; (void)ws_size;
  const float* x      = (const float*)d_in[0];
  const float* c1_w   = (const float*)d_in[1];
  const float* c1_b   = (const float*)d_in[2];
  const float* cd_w   = (const float*)d_in[3];
  const float* cd_b   = (const float*)d_in[4];
  const float* bn1_g  = (const float*)d_in[5];
  const float* bn1_b  = (const float*)d_in[6];
  const float* h_qkv_w= (const float*)d_in[7];
  const float* h_bq_g = (const float*)d_in[8];
  const float* h_bq_b = (const float*)d_in[9];
  const float* h_bs_g = (const float*)d_in[10];
  const float* h_bs_b = (const float*)d_in[11];
  const float* h_bo_g = (const float*)d_in[12];
  const float* h_bo_b = (const float*)d_in[13];
  const float* h_rel  = (const float*)d_in[14];
  const float* w_qkv_w= (const float*)d_in[15];
  const float* w_bq_g = (const float*)d_in[16];
  const float* w_bq_b = (const float*)d_in[17];
  const float* w_bs_g = (const float*)d_in[18];
  const float* w_bs_b = (const float*)d_in[19];
  const float* w_bo_g = (const float*)d_in[20];
  const float* w_bo_b = (const float*)d_in[21];
  const float* w_rel  = (const float*)d_in[22];
  const float* cu_w   = (const float*)d_in[23];
  const float* cu_b   = (const float*)d_in[24];
  const float* bn2_g  = (const float*)d_in[25];
  const float* bn2_b  = (const float*)d_in[26];
  float* out = (float*)d_out;

  char* ws = (char*)d_ws;
  const size_t MB = 1ull << 20;
  unsigned short* xbf = (unsigned short*)(ws);              // 16 MB  (x, K-major)
  unsigned short* gin = (unsigned short*)(ws + 16 * MB);    // 16 MB  (stage input, K-major)
  unsigned short* wbf = (unsigned short*)(ws + 32 * MB);    // weights (<0.5 MB)
  unsigned short* w_c1 = wbf;
  unsigned short* w_cd = wbf + 32768;
  unsigned short* w_hq = wbf + 49152;
  unsigned short* w_wq = wbf + 81920;
  unsigned short* w_cu = wbf + 114688;
  float* stats  = (float*)(ws + 32 * MB + (512u << 10));    // control area
  float* ss_bn1 = stats + 2048;
  float* ss_bq  = stats + 2560;
  float* ss_sim = stats + 3072;
  float* ss_bo  = stats + 3584;
  float* ss_bn2 = stats + 4096;
  float* xout   = (float*)(ws + 33 * MB);                   // 64 MB
  float* gout   = (float*)(ws + 97 * MB);                   // 64 MB
  float* attout = (float*)(ws + 161 * MB);                  // 64 MB

  const float inv65536 = 1.0f / 65536.0f;
  const float invSim   = 1.0f / 4194304.0f;

  // conversions: weights to bf16, x to K-major bf16
  k_xT<<<8192, 256, 0, stream>>>(x, xbf);
  k_f32_to_bf16<<<128, 256, 0, stream>>>(c1_w, w_c1, COUT_ * CIN_);
  k_f32_to_bf16<<<64, 256, 0, stream>>>(cd_w, w_cd, MID_ * CIN_);
  k_f32_to_bf16<<<128, 256, 0, stream>>>(h_qkv_w, w_hq, 256 * 128);
  k_f32_to_bf16<<<128, 256, 0, stream>>>(w_qkv_w, w_wq, 256 * 128);
  k_f32_to_bf16<<<128, 256, 0, stream>>>(cu_w, w_cu, 256 * 128);

  // c1: x_out = c1_w @ x + c1_b   (M=256, ncols=65536 -> 16*1024 waves)
  k_gemm_bf16<<<4096, 128, 0, stream>>>(w_c1, xbf, c1_b, xout, 256, 128, HW_);
  // cd: mid = cd_w @ x + cd_b     (M=128 -> 8*1024 waves)
  k_gemm_bf16<<<2048, 128, 0, stream>>>(w_cd, xbf, cd_b, gout, 128, 128, HW_);
  // bn1 + relu -> h_in (bf16, K-major)
  k_stats<<<128, 256, 0, stream>>>(gout, stats, 128, NB_, HW_);
  k_bn_finalize<<<1, 256, 0, stream>>>(stats, bn1_g, bn1_b, ss_bn1, 128, inv65536);
  k_bn1_relu_to_hin<<<32768, 256, 0, stream>>>(gout, ss_bn1, gin);

  // ---- height axial ----
  k_gemm_bf16<<<4096, 128, 0, stream>>>(w_hq, gin, nullptr, gout, 256, 128, 64);
  k_stats<<<256, 256, 0, stream>>>(gout, stats, 256, SEQB_, 64);
  k_bn_finalize<<<1, 256, 0, stream>>>(stats, h_bq_g, h_bq_b, ss_bq, 256, inv65536);
  k_zero<<<1, 64, 0, stream>>>(stats, 64);
  k_att_stats<<<SEQB_ * G_, 256, 0, stream>>>(gout, ss_bq, h_rel, stats);
  k_bn_finalize<<<1, 32, 0, stream>>>(stats, h_bs_g, h_bs_b, ss_sim, 24, invSim);
  k_att_apply<<<SEQB_ * G_, 64, 0, stream>>>(gout, ss_bq, h_rel, ss_sim, attout);
  k_stats<<<256, 256, 0, stream>>>(attout, stats, 256, SEQB_, 64);
  k_bn_finalize<<<1, 256, 0, stream>>>(stats, h_bo_g, h_bo_b, ss_bo, 256, inv65536);
  k_bo_apply_h<<<32768, 256, 0, stream>>>(attout, ss_bo, gin);

  // ---- width axial ----
  k_gemm_bf16<<<4096, 128, 0, stream>>>(w_wq, gin, nullptr, gout, 256, 128, 64);
  k_stats<<<256, 256, 0, stream>>>(gout, stats, 256, SEQB_, 64);
  k_bn_finalize<<<1, 256, 0, stream>>>(stats, w_bq_g, w_bq_b, ss_bq, 256, inv65536);
  k_zero<<<1, 64, 0, stream>>>(stats, 64);
  k_att_stats<<<SEQB_ * G_, 256, 0, stream>>>(gout, ss_bq, w_rel, stats);
  k_bn_finalize<<<1, 32, 0, stream>>>(stats, w_bs_g, w_bs_b, ss_sim, 24, invSim);
  k_att_apply<<<SEQB_ * G_, 64, 0, stream>>>(gout, ss_bq, w_rel, ss_sim, attout);
  k_stats<<<256, 256, 0, stream>>>(attout, stats, 256, SEQB_, 64);
  k_bn_finalize<<<1, 256, 0, stream>>>(stats, w_bo_g, w_bo_b, ss_bo, 256, inv65536);
  k_bo_apply_w<<<32768, 256, 0, stream>>>(attout, ss_bo, gin);

  // ---- cu + bn2 + residual ----
  k_gemm_bf16<<<4096, 128, 0, stream>>>(w_cu, gin, cu_b, gout, 256, 128, HW_);
  k_stats<<<256, 256, 0, stream>>>(gout, stats, 256, NB_, HW_);
  k_bn_finalize<<<1, 256, 0, stream>>>(stats, bn2_g, bn2_b, ss_bn2, 256, inv65536);
  k_bn2_res<<<65536, 256, 0, stream>>>(gout, ss_bn2, xout, out);
}
